// Model_3573412790429
// MI455X (gfx1250) — matrix operand order repaired and checked by reading, not compiled, not run on hardware
//
#include <hip/hip_runtime.h>
#include <hip/hip_bf16.h>

typedef _Float16 v16h __attribute__((ext_vector_type(16)));
typedef _Float16 v8h  __attribute__((ext_vector_type(8)));
typedef float    v8f  __attribute__((ext_vector_type(8)));

union AF { v16h v; v8h h2[2]; _Float16 e[16]; };

// Same-wave LDS ordering fence: LDS ops from one wave execute in order; this
// forces completion + stops the compiler from moving LDS ops across it.
__device__ __forceinline__ void lds_fence() { asm volatile("s_wait_dscnt 0" ::: "memory"); }

// DPP16-based add of a permuted copy (VALU pipe, no LDS traffic).
template<int CTRL>
__device__ __forceinline__ float dpp_add(float v) {
    int y = __builtin_amdgcn_update_dpp(0, __float_as_int(v), CTRL, 0xf, 0xf, true);
    return v + __int_as_float(y);
}
// Sum across each 16-lane half. quad_perm xor1, xor2, then half_mirror/mirror
// (equivalent to xor4/xor8 once values are uniform within quads/octs).
__device__ __forceinline__ float rowsum16(float v) {
    v = dpp_add<0xB1>(v);   // quad_perm [1,0,3,2]
    v = dpp_add<0x4E>(v);   // quad_perm [2,3,0,1]
    v = dpp_add<0x141>(v);  // row_half_mirror
    v = dpp_add<0x140>(v);  // row_mirror
    return v;
}

struct Params {
    const float *x;
    const float *W0,*b0,*W1,*b1,*W2,*b2,*W3,*b3,*W4,*b4,*W5,*b5,*W6,*b6;
    const float *g0,*be0,*g1,*be1,*g2,*be2,*g3,*be3,*g4,*be4,*g5,*be5;
    float *out;
    int nrows;
};

// Dual-tile WMMA layer: two independent 16-row tiles (H0, H1) share every
// B-fragment and param load; two WMMA/LN chains give the scheduler work to
// fill WMMA->VALU hazard slots and DPP chain latency.
// A frag (16-bit, 16x32): lane<16: K = kb+{0..7}, kb+16+{0..7}; lane>=16: +8.
// B frag (16-bit, 32x16): lane<16: K = kb+{0..7}, kb+16+{0..7}; lane>=16: +8; N = lane&15.
// C/D: VGPR r -> row (r + 8*(lane>=16)), col = lane&15.
// sPf[(POFF+n)*4] = {gamma, beta, bias, 0}.
template<int K, int NOUT, int WOFF, int POFF>
__device__ __forceinline__ void wmma_layer2(_Float16* H0, _Float16* H1,
                                            const _Float16* sWt, const float* sPf,
                                            int lane)
{
    constexpr int KC = K / 32;
    constexpr int NT = NOUT / 16;
    const int half = lane >> 4;
    const int nl   = lane & 15;
    // ---- A fragments (activations, both tiles) ----
    AF a0[KC], a1[KC];
#pragma unroll
    for (int c = 0; c < KC; ++c) {
        const int kb = c * 32;
        a0[c].h2[0] = *(const v8h*)&H0[nl * 128 + kb + half * 8];
        a0[c].h2[1] = *(const v8h*)&H0[nl * 128 + kb + 16 + half * 8];
        a1[c].h2[0] = *(const v8h*)&H1[nl * 128 + kb + half * 8];
        a1[c].h2[1] = *(const v8h*)&H1[nl * 128 + kb + 16 + half * 8];
    }
    // ---- matmuls: B fragment loaded once, used by both tiles ----
    v8f acc0[NT], acc1[NT];
#pragma unroll
    for (int t = 0; t < NT; ++t) {
#pragma unroll
        for (int r = 0; r < 8; ++r) { acc0[t][r] = 0.0f; acc1[t][r] = 0.0f; }
#pragma unroll
        for (int c = 0; c < KC; ++c) {
            AF b;
            const int base = WOFF + (t * 16 + nl) * K + c * 32 + half * 8;
            b.h2[0] = *(const v8h*)&sWt[base];
            b.h2[1] = *(const v8h*)&sWt[base + 16];
            acc0[t] = __builtin_amdgcn_wmma_f32_16x16x32_f16(
                false, a0[c].v, false, b.v, (short)0, acc0[t], false, false);
            acc1[t] = __builtin_amdgcn_wmma_f32_16x16x32_f16(
                false, a1[c].v, false, b.v, (short)0, acc1[t], false, false);
        }
        const float bias = sPf[(POFF + t * 16 + nl) * 4 + 2];
#pragma unroll
        for (int r = 0; r < 8; ++r) { acc0[t][r] += bias; acc1[t][r] += bias; }
    }
    // ---- LayerNorm + ReLU for both tiles (DPP reductions, two chains) ----
    float s1a[8], s2a[8], s1b[8], s2b[8];
#pragma unroll
    for (int r = 0; r < 8; ++r) {
        s1a[r] = 0.0f; s2a[r] = 0.0f; s1b[r] = 0.0f; s2b[r] = 0.0f;
#pragma unroll
        for (int t = 0; t < NT; ++t) {
            float v0 = acc0[t][r]; s1a[r] += v0; s2a[r] += v0 * v0;
            float v1 = acc1[t][r]; s1b[r] += v1; s2b[r] += v1 * v1;
        }
        s1a[r] = rowsum16(s1a[r]); s2a[r] = rowsum16(s2a[r]);
        s1b[r] = rowsum16(s1b[r]); s2b[r] = rowsum16(s2b[r]);
    }
    const float invF = 1.0f / (float)NOUT;
    float mean0[8], rs0[8], mean1[8], rs1[8];
#pragma unroll
    for (int r = 0; r < 8; ++r) {
        mean0[r] = s1a[r] * invF;
        rs0[r] = rsqrtf(s2a[r] * invF - mean0[r] * mean0[r] + 1e-5f);
        mean1[r] = s1b[r] * invF;
        rs1[r] = rsqrtf(s2b[r] * invF - mean1[r] * mean1[r] + 1e-5f);
    }
#pragma unroll
    for (int t = 0; t < NT; ++t) {
        const int n = t * 16 + nl;
        const float2 gb = *(const float2*)&sPf[(POFF + n) * 4];
#pragma unroll
        for (int r = 0; r < 8; ++r) {
            float v0 = (acc0[t][r] - mean0[r]) * rs0[r] * gb.x + gb.y;
            float v1 = (acc1[t][r] - mean1[r]) * rs1[r] * gb.x + gb.y;
            H0[(r + half * 8) * 128 + n] = (_Float16)fmaxf(v0, 0.0f);
            H1[(r + half * 8) * 128 + n] = (_Float16)fmaxf(v1, 0.0f);
        }
    }
}

__global__ __launch_bounds__(64) void mlp7_wmma(Params p)
{
    // ---- static LDS (~63.4 KB) ----
    __shared__ _Float16 sWt[21504];      // Wt concat: W1(pad 32x32)@0, W2@1024, W3@3072, W4@11264, W5@19456
    __shared__ float sPf[1280];          // {g,be,bias,0} per n, layers 1..5 @ n-offsets {0,32,96,224,288}
    __shared__ float sW0[16], sB0[16], sG0[16], sBe0[16], sW6[32];
    __shared__ _Float16 sH[2][2][16 * 128]; // [wave][tile] activation buffers

    const int tid = threadIdx.x;

    // ---- stage transposed f16 weights ----
    for (int i = tid; i < 1024; i += 64) { int n = i >> 5, k = i & 31;
        sWt[i] = (_Float16)(k < 16 ? p.W1[k * 32 + n] : 0.0f); }           // 16->32, pad K to 32
    for (int i = tid; i < 2048; i += 64) { int n = i >> 5, k = i & 31;
        sWt[1024 + i] = (_Float16)p.W2[k * 64 + n]; }                      // 32->64
    for (int i = tid; i < 8192; i += 64) { int n = i >> 6, k = i & 63;
        sWt[3072 + i] = (_Float16)p.W3[k * 128 + n]; }                     // 64->128
    for (int i = tid; i < 8192; i += 64) { int n = i >> 7, k = i & 127;
        sWt[11264 + i] = (_Float16)p.W4[k * 64 + n]; }                     // 128->64
    for (int i = tid; i < 2048; i += 64) { int n = i >> 6, k = i & 63;
        sWt[19456 + i] = (_Float16)p.W5[k * 32 + n]; }                     // 64->32

    { const float* bs[5]  = {p.b1, p.b2, p.b3, p.b4, p.b5};
      const float* gs[5]  = {p.g1, p.g2, p.g3, p.g4, p.g5};
      const float* bes[5] = {p.be1, p.be2, p.be3, p.be4, p.be5};
      const int bd[5] = {32, 64, 128, 64, 32};
      const int bo[5] = {0, 32, 96, 224, 288};
      for (int l = 0; l < 5; ++l)
          for (int i = tid; i < bd[l]; i += 64) {
              const int b4 = (bo[l] + i) * 4;
              sPf[b4 + 0] = gs[l][i]; sPf[b4 + 1] = bes[l][i];
              sPf[b4 + 2] = bs[l][i]; sPf[b4 + 3] = 0.0f; } }
    for (int i = tid; i < 16; i += 64) {
        sW0[i] = p.W0[i]; sB0[i] = p.b0[i]; sG0[i] = p.g0[i]; sBe0[i] = p.be0[i]; }
    for (int i = tid; i < 32; i += 64) sW6[i] = p.W6[i];
    __syncthreads();

    const int wave = tid >> 5, lane = tid & 31;
    const int half = lane >> 4, m = lane & 15;
    _Float16* H0 = sH[wave][0];
    _Float16* H1 = sH[wave][1];
    const float b6v = p.b6[0];

    // ---- layer0 closed-form LN precompute ----
    // z_j = x*w_j + b_j ; z-mean = x*dw_j + db_j ; var = x^2*Sww + 2x*Swb + Sbb
    float mw = 0.0f, mb = 0.0f;
    for (int c = 0; c < 16; ++c) { mw += sW0[c]; mb += sB0[c]; }
    mw *= (1.0f / 16.0f); mb *= (1.0f / 16.0f);
    float Sww = 0.0f, Swb = 0.0f, Sbb = 0.0f;
    for (int c = 0; c < 16; ++c) {
        float dw = sW0[c] - mw, db = sB0[c] - mb;
        Sww += dw * dw; Swb += dw * db; Sbb += db * db;
    }
    Sww *= (1.0f / 16.0f); Swb *= (2.0f / 16.0f); Sbb = Sbb * (1.0f / 16.0f) + 1e-5f;
    float A0[8], B0[8], E0[8];    // h_j = relu(rs*(x*A0+B0) + E0)
#pragma unroll
    for (int j = 0; j < 8; ++j) {
        int c = half * 8 + j;
        float dw = sW0[c] - mw, db = sB0[c] - mb, g = sG0[c];
        A0[j] = dw * g; B0[j] = db * g; E0[j] = sBe0[c];
    }

    const int nsup = (p.nrows + 31) >> 5;   // 32-row supertiles
    const int gw = blockIdx.x * 2 + wave;
    const int nw = gridDim.x * 2;

    for (int sup = gw; sup < nsup; sup += nw) {
        const long base = (long)sup << 5;
        const long row0 = base + m;
        const long row1 = base + 16 + m;

        // ---- layer0: 1->16, LN0 + ReLU (analytic, both tiles) ----
        float xv0 = (row0 < (long)p.nrows) ? p.x[row0] : 0.0f;
        float xv1 = (row1 < (long)p.nrows) ? p.x[row1] : 0.0f;
        const float r0 = rsqrtf(xv0 * xv0 * Sww + xv0 * Swb + Sbb);
        const float r1 = rsqrtf(xv1 * xv1 * Sww + xv1 * Swb + Sbb);
        lds_fence();  // prior supertile's layer6 reads done before overwrite
#pragma unroll
        for (int j = 0; j < 8; ++j) {
            float v0 = r0 * (xv0 * A0[j] + B0[j]) + E0[j];
            float v1 = r1 * (xv1 * A0[j] + B0[j]) + E0[j];
            H0[m * 128 + half * 8 + j] = (_Float16)fmaxf(v0, 0.0f);
            H1[m * 128 + half * 8 + j] = (_Float16)fmaxf(v1, 0.0f);
        }
        { AF z;                                  // zero pad cols 16..31 (layer1 K padded to 32)
#pragma unroll
          for (int j = 0; j < 8; ++j) z.e[j] = (_Float16)0.0f;
          *(v8h*)&H0[m * 128 + 16 + half * 8] = z.h2[0];
          *(v8h*)&H1[m * 128 + 16 + half * 8] = z.h2[0]; }
        lds_fence();

        // ---- layers 1..5: dual-tile WMMA + fused LN + ReLU ----
        wmma_layer2<32,  32,     0,   0>(H0, H1, sWt, sPf, lane); lds_fence();
        wmma_layer2<32,  64,  1024,  32>(H0, H1, sWt, sPf, lane); lds_fence();
        wmma_layer2<64, 128,  3072,  96>(H0, H1, sWt, sPf, lane); lds_fence();
        wmma_layer2<128, 64, 11264, 224>(H0, H1, sWt, sPf, lane); lds_fence();
        wmma_layer2<64,  32, 19456, 288>(H0, H1, sWt, sPf, lane); lds_fence();

        // ---- layer6: 32->1 dot (vectorized LDS reads) ----
        float4 w6v[4];
#pragma unroll
        for (int q = 0; q < 4; ++q) w6v[q] = *(const float4*)&sW6[half * 16 + q * 4];
        AF hv0, hv1;
        hv0.h2[0] = *(const v8h*)&H0[m * 128 + half * 16];
        hv0.h2[1] = *(const v8h*)&H0[m * 128 + half * 16 + 8];
        hv1.h2[0] = *(const v8h*)&H1[m * 128 + half * 16];
        hv1.h2[1] = *(const v8h*)&H1[m * 128 + half * 16 + 8];
        float s0 = 0.0f, s1 = 0.0f;
#pragma unroll
        for (int q = 0; q < 4; ++q) {
            const float* wq = (const float*)&w6v[q];
#pragma unroll
            for (int e = 0; e < 4; ++e) {
                s0 += (float)hv0.e[q * 4 + e] * wq[e];
                s1 += (float)hv1.e[q * 4 + e] * wq[e];
            }
        }
        s0 += __shfl_xor(s0, 16, 32);   // combine lane halves (crosses DPP16 row)
        s1 += __shfl_xor(s1, 16, 32);
        if (half == 0 && row0 < (long)p.nrows) p.out[row0] = s0 + b6v;
        if (half == 0 && row1 < (long)p.nrows) p.out[row1] = s1 + b6v;
    }
}

extern "C" void kernel_launch(void* const* d_in, const int* in_sizes, int n_in,
                              void* d_out, int out_size, void* d_ws, size_t ws_size,
                              hipStream_t stream)
{
    Params p;
    p.x  = (const float*)d_in[0];
    p.W0 = (const float*)d_in[1];  p.b0 = (const float*)d_in[2];
    p.W1 = (const float*)d_in[3];  p.b1 = (const float*)d_in[4];
    p.W2 = (const float*)d_in[5];  p.b2 = (const float*)d_in[6];
    p.W3 = (const float*)d_in[7];  p.b3 = (const float*)d_in[8];
    p.W4 = (const float*)d_in[9];  p.b4 = (const float*)d_in[10];
    p.W5 = (const float*)d_in[11]; p.b5 = (const float*)d_in[12];
    p.W6 = (const float*)d_in[13]; p.b6 = (const float*)d_in[14];
    p.g0 = (const float*)d_in[15]; p.be0 = (const float*)d_in[16];
    p.g1 = (const float*)d_in[17]; p.be1 = (const float*)d_in[18];
    p.g2 = (const float*)d_in[19]; p.be2 = (const float*)d_in[20];
    p.g3 = (const float*)d_in[21]; p.be3 = (const float*)d_in[22];
    p.g4 = (const float*)d_in[23]; p.be4 = (const float*)d_in[24];
    p.g5 = (const float*)d_in[25]; p.be5 = (const float*)d_in[26];
    p.out = (float*)d_out;
    p.nrows = in_sizes[0];

    const int nsup = (p.nrows + 31) / 32;
    int maxb = (nsup + 1) / 2;            // 2 waves per block, 1 supertile per wave minimum
    if (maxb < 1) maxb = 1;
    int blocks = maxb < 4096 ? maxb : 4096;
    mlp7_wmma<<<blocks, 64, 0, stream>>>(p);
}
